// NoGraphLayer_16088947490886
// MI455X (gfx1250) — matrix operand order repaired and checked by reading, not compiled, not run on hardware
//
#include <hip/hip_runtime.h>
#include <hip/hip_bf16.h>
#include <stdint.h>

// Problem sizes (compile-time constants, match reference)
#define BB 32768
#define KN 32
#define DD 256
#define HH 8
#define DKH 32
#define FFD 1024
#define S_PAD 264   // padded LDS row stride in bf16 elements (keeps 16B align, breaks bank conflicts)
#define GPAD 1032   // padded stride for the [16][1024] gelu buffer

typedef __attribute__((ext_vector_type(16))) __bf16 v16bf;
typedef __attribute__((ext_vector_type(8)))  float  v8f;
typedef __attribute__((ext_vector_type(4)))  int    v4i;

// gfx1250 async global->LDS copy: only if toolchain exposes both the copy and the wait
#if defined(__has_builtin)
#if __has_builtin(__builtin_amdgcn_global_load_async_to_lds_b128) && \
    __has_builtin(__builtin_amdgcn_s_wait_asynccnt)
#define HAVE_ASYNC_LDS 1
typedef __attribute__((address_space(1))) v4i* as1_v4i_ptr;
typedef __attribute__((address_space(3))) v4i* as3_v4i_ptr;
#endif
#endif

union Frag { uint4 q[2]; v16bf v; uint32_t u[8]; };

__device__ inline uint16_t f2bf(float x) {
    union { __bf16 h; uint16_t u; } cv;
    cv.h = (__bf16)x;
    return cv.u;
}

// ---- WMMA fragment loaders (wave32, bf16 16x16x32 fragments, contiguous k = 16*half + s in A and B) ----

// A matrix 16x32 bf16: lane L -> m = L%16, half = L/16.
// VGPR v0..7 hold K = ks*32 + half*16 + {0..15} (contiguous)
__device__ inline v16bf load_afrag(const uint16_t* base, int rowStride, int ks, int validRows) {
    int lane = threadIdx.x & 31;
    int m = lane & 15, half = lane >> 4;
    Frag f;
    if (m < validRows) {
        const uint16_t* p = base + m * rowStride + ks * 32 + half * 16;
        f.q[0] = *(const uint4*)(p);
        f.q[1] = *(const uint4*)(p + 8);
    } else {
        f.q[0] = make_uint4(0u, 0u, 0u, 0u);
        f.q[1] = make_uint4(0u, 0u, 0u, 0u);
    }
    return f.v;
}

// B matrix 32x16 bf16, LDS stored n-major [n][k]: lane L -> n = L%16, half = L/16.
// VGPR v holds K pair (half*16 + 2v, half*16 + 2v + 1): 16 contiguous bf16 at k0 = ks*32 + half*16
__device__ inline v16bf load_bfrag(const uint16_t* base, int rowStride, int ks) {
    int lane = threadIdx.x & 31;
    int n = lane & 15, half = lane >> 4;
    const uint16_t* p = base + n * rowStride + ks * 32 + half * 16;
    Frag f;
    f.q[0] = *(const uint4*)(p);
    f.q[1] = *(const uint4*)(p + 8);
    return f.v;
}

__device__ inline v8f wmma_bf16(v16bf a, v16bf b, v8f c) {
    return __builtin_amdgcn_wmma_f32_16x16x32_bf16(false, a, false, b, (short)0, c, false, false);
}

// C/D 16x16 f32: lane L -> n = L%16; VGPR v -> m = v + 8*(L/16)
__device__ inline void store_c_bf16(uint16_t* base, int rowStride, v8f c, int validRows) {
    int lane = threadIdx.x & 31;
    int n = lane & 15, half = lane >> 4;
#pragma unroll
    for (int v = 0; v < 8; ++v) {
        int m = v + 8 * half;
        if (m < validRows) base[m * rowStride + n] = f2bf(c[v]);
    }
}

// 16xN GEMM inner loop: A row-major [m][k] in LDS, B n-major [n][k] in LDS
__device__ inline v8f gemm16(const uint16_t* A, int lda, int arows,
                             const uint16_t* Bn, int ldb, int ksteps, v8f acc) {
#pragma unroll
    for (int ks = 0; ks < ksteps; ++ks) {
        v16bf a = load_afrag(A, lda, ks, arows);
        v16bf b = load_bfrag(Bn, ldb, ks);
        acc = wmma_bf16(a, b, acc);
    }
    return acc;
}

// Stage W^T chunk into LDS n-major: lds[nloc][k] = W[k][c0+nloc], k in [0,256), nloc in [0,128)
__device__ inline void stageT(uint16_t* lds, const float* __restrict__ W, int ldw, int c0) {
    int t = threadIdx.x;
    for (int base = t * 4; base < 256 * 128; base += 256 * 4) {
        int k  = base >> 7;
        int nl = base & 127;
        float4 wv = *(const float4*)(W + (size_t)k * ldw + c0 + nl);
        lds[(nl + 0) * S_PAD + k] = f2bf(wv.x);
        lds[(nl + 1) * S_PAD + k] = f2bf(wv.y);
        lds[(nl + 2) * S_PAD + k] = f2bf(wv.z);
        lds[(nl + 3) * S_PAD + k] = f2bf(wv.w);
    }
}

// Stage W rows directly (no transpose): lds[rloc][c] = W[r0+rloc][c], c in [0,256), rloc in [0,128)
__device__ inline void stageD(uint16_t* lds, const float* __restrict__ W, int ldw, int r0) {
    int t = threadIdx.x;
    for (int base = t * 4; base < 128 * 256; base += 256 * 4) {
        int r = base >> 8;
        int c = base & 255;
        float4 wv = *(const float4*)(W + (size_t)(r0 + r) * ldw + c);
        uint16_t* p = lds + r * S_PAD + c;
        p[0] = f2bf(wv.x); p[1] = f2bf(wv.y); p[2] = f2bf(wv.z); p[3] = f2bf(wv.w);
    }
}

// LayerNorm of one 256-wide row, 16 lanes per row (lane handles 16 contiguous cols)
__device__ inline void ln_row(const float* __restrict__ xrow,
                              const float* __restrict__ g, const float* __restrict__ bt,
                              uint16_t* __restrict__ dst, float* __restrict__ dstF) {
    int lane = threadIdx.x & 31;
    int c0 = (lane & 15) * 16;
    float vv[16];
#pragma unroll
    for (int j = 0; j < 16; j += 4) {
        float4 t4 = *(const float4*)(xrow + c0 + j);
        vv[j] = t4.x; vv[j + 1] = t4.y; vv[j + 2] = t4.z; vv[j + 3] = t4.w;
    }
    float s = 0.f, s2 = 0.f;
#pragma unroll
    for (int j = 0; j < 16; ++j) { s += vv[j]; s2 += vv[j] * vv[j]; }
#pragma unroll
    for (int msk = 8; msk >= 1; msk >>= 1) {
        s  += __shfl_xor(s, msk, 32);
        s2 += __shfl_xor(s2, msk, 32);
    }
    float mu = s * (1.0f / 256.0f);
    float var = s2 * (1.0f / 256.0f) - mu * mu;
    float rs = rsqrtf(var + 1e-5f);
#pragma unroll
    for (int j = 0; j < 16; ++j) {
        float y = (vv[j] - mu) * rs * g[c0 + j] + bt[c0 + j];
        dst[c0 + j] = f2bf(y);
        if (dstF) dstF[c0 + j] = vv[j];
    }
}

// Prefetch a contiguous f32 region into cache (global_prefetch_b8), one 128B line per lane-step
__device__ inline void prefetch_f32(const float* p, int nfloats) {
    int lane = threadIdx.x & 31;
    for (int off = lane * 32; off < nfloats; off += 32 * 32)
        __builtin_prefetch(p + off, 0, 1);
}

// =====================================================================================
// Kernel A: fused LN1 + Q + Qhat(Q pushed through W_k) + scores + softmax + ctx +
//           ctx@W_v (head) + @W_o + residual  -> x  (per 16-anchor tile, 8 waves)
// =====================================================================================
__global__ void __launch_bounds__(256)
attn_kernel(const float* __restrict__ xa, const float* __restrict__ xng,
            const float* __restrict__ Wq, const float* __restrict__ Wk,
            const float* __restrict__ Wv, const float* __restrict__ Wo,
            const float* __restrict__ g1, const float* __restrict__ b1,
            float* __restrict__ xout) {
    extern __shared__ char smem[];
    uint16_t* lnx   = (uint16_t*)smem;                  // [16][S_PAD] ln(x_anc), later attnout
    uint16_t* Qbf   = lnx + 16 * S_PAD;                 // [16][S_PAD] Q
    uint16_t* qc    = Qbf + 16 * S_PAD;                 // [16][8][S_PAD] Qhat, then ctx (per anchor)
    uint16_t* attnb = qc + 16 * 8 * S_PAD;              // [8 waves][16][40] attn weights
    uint16_t* uni   = attnb + 8 * 16 * 40;              // union: panel [128][S_PAD]  /  xn [8][32][S_PAD]
    uint16_t* panel = uni;

    int b0 = blockIdx.x * 16;
    int tid = threadIdx.x, lane = tid & 31, w = tid >> 5;

    // ---- Phase 0: LN1 (wave w handles rows 2w, 2w+1) + zero attn buffer pad rows ----
    {
        int r = 2 * w + (lane >> 4);
        ln_row(xa + (size_t)(b0 + r) * DD, g1, b1, lnx + r * S_PAD, nullptr);
    }
    for (int i = tid; i < 8 * 16 * 40; i += 256) attnb[i] = 0;
    // pull this wave's round-0 neighbor tile toward L2/WGP$ while GEMM phases run
    prefetch_f32(xng + ((size_t)(b0 + w) * KN) * DD, KN * DD);
    __syncthreads();

    // ---- Phase 1: Q = lnx @ W_q  (M=16, N=256 in two 128-col chunks, K=256) ----
#pragma unroll
    for (int ch = 0; ch < 2; ++ch) {
        stageT(panel, Wq, DD, ch * 128);
        __syncthreads();
        v8f acc = {0.f, 0.f, 0.f, 0.f, 0.f, 0.f, 0.f, 0.f};
        acc = gemm16(lnx, S_PAD, 16, panel + (w * 16) * S_PAD, S_PAD, 8, acc);
        store_c_bf16(Qbf + ch * 128 + w * 16, S_PAD, acc, 16);
        __syncthreads();
    }

    // ---- Phase 2: Qhat[b,h,d'] = sum_dk Q[b,h*32+dk] * W_k[d'][h*32+dk] ----
    // B stays in W_k's natural row-major layout (n = d', k = dk contiguous)
#pragma unroll
    for (int ch = 0; ch < 2; ++ch) {
        stageD(panel, Wk, DD, ch * 128);
        __syncthreads();
#pragma unroll
        for (int h = 0; h < HH; ++h) {
            v8f acc = {0.f, 0.f, 0.f, 0.f, 0.f, 0.f, 0.f, 0.f};
            v16bf a = load_afrag(Qbf, S_PAD, h, 16);                      // k-base = h*32
            v16bf b = load_bfrag(panel + (w * 16) * S_PAD, S_PAD, h);     // k-base = h*32
            acc = wmma_bf16(a, b, acc);
            store_c_bf16(qc + h * S_PAD + ch * 128 + w * 16, 8 * S_PAD, acc, 16);
        }
        __syncthreads();
    }

    // ---- Phase 3: per-anchor scores + softmax + ctx (one anchor per wave, 2 rounds) ----
    uint16_t* xn = uni;  // reuse panel memory: barriers above guarantee safety
    const float scale = 0.17677669529663689f;  // 1/sqrt(32)
#pragma unroll
    for (int rd = 0; rd < 2; ++rd) {
        int a = rd * 8 + w;
        uint16_t* xw = xn + w * (32 * S_PAD);
        // stage x_nei[b0+a] : [32][256] f32 -> bf16 (wave-cooperative, read ONCE from HBM)
        const float* src = xng + ((size_t)(b0 + a) * KN) * DD;
        for (int i = lane * 4; i < KN * DD; i += 32 * 4) {
            int r = i >> 8, c = i & 255;
            float4 t4 = *(const float4*)(src + i);
            uint16_t* p = xw + r * S_PAD + c;
            p[0] = f2bf(t4.x); p[1] = f2bf(t4.y); p[2] = f2bf(t4.z); p[3] = f2bf(t4.w);
        }
        // overlap: prefetch this wave's round-1 tile while round-0 math runs
        if (rd == 0)
            prefetch_f32(xng + ((size_t)(b0 + 8 + w) * KN) * DD, KN * DD);
        // scores[h, nb] = Qhat[a,h,:] . x_nei[nb,:]   (A rows = 8 heads, B = xw n-major natural)
        const uint16_t* Aq = qc + (size_t)a * 8 * S_PAD;
        v8f s0 = {0.f, 0.f, 0.f, 0.f, 0.f, 0.f, 0.f, 0.f};
        v8f s1 = {0.f, 0.f, 0.f, 0.f, 0.f, 0.f, 0.f, 0.f};
        s0 = gemm16(Aq, S_PAD, 8, xw, S_PAD, 8, s0);
        s1 = gemm16(Aq, S_PAD, 8, xw + 16 * S_PAD, S_PAD, 8, s1);
        // softmax across 32 neighbors; valid heads live in half==0 lanes (m = v)
        float at0[8], at1[8];
#pragma unroll
        for (int v = 0; v < 8; ++v) {
            float x0 = s0[v] * scale, x1 = s1[v] * scale;
            float mx = fmaxf(x0, x1);
#pragma unroll
            for (int msk = 8; msk >= 1; msk >>= 1) mx = fmaxf(mx, __shfl_xor(mx, msk, 32));
            float e0 = __expf(x0 - mx), e1 = __expf(x1 - mx);
            float sm = e0 + e1;
#pragma unroll
            for (int msk = 8; msk >= 1; msk >>= 1) sm += __shfl_xor(sm, msk, 32);
            float inv = 1.0f / sm;
            at0[v] = e0 * inv; at1[v] = e1 * inv;
        }
        uint16_t* ab = attnb + w * 16 * 40;
        if (lane < 16) {
#pragma unroll
            for (int v = 0; v < 8; ++v) {
                ab[v * 40 + lane]      = f2bf(at0[v]);
                ab[v * 40 + 16 + lane] = f2bf(at1[v]);
            }
        }
        // ctx[h, d'] = sum_nb attn[h,nb] * x_nei[nb,d']  (K = 32 neighbors, one WMMA step)
        v16bf afr = load_afrag(ab, 40, 0, 16);
        int n = lane & 15, half = lane >> 4;
#pragma unroll
        for (int nt = 0; nt < 16; ++nt) {
            Frag bf;
#pragma unroll
            for (int v = 0; v < 8; ++v) {
                int k = half * 16 + 2 * v;
                uint32_t lo = xw[k * S_PAD + nt * 16 + n];
                uint32_t hi = xw[(k + 1) * S_PAD + nt * 16 + n];
                bf.u[v] = lo | (hi << 16);
            }
            v8f c = {0.f, 0.f, 0.f, 0.f, 0.f, 0.f, 0.f, 0.f};
            c = wmma_bf16(afr, bf.v, c);
            store_c_bf16(qc + (size_t)a * 8 * S_PAD + nt * 16, S_PAD, c, 8);  // overwrite Qhat[a] with ctx[a]
        }
    }
    __syncthreads();

    // ---- Phase 4: attnout[b, h*32+dk] = ctx[b,h,:] @ W_v[:, h*32+dk] ----
    uint16_t* attnout = lnx;  // reuse
#pragma unroll
    for (int ch = 0; ch < 2; ++ch) {
        stageT(panel, Wv, DD, ch * 128);
        __syncthreads();
        int cg = ch * 128 + w * 16;
        int h = cg >> 5;
        v8f acc = {0.f, 0.f, 0.f, 0.f, 0.f, 0.f, 0.f, 0.f};
        acc = gemm16(qc + h * S_PAD, 8 * S_PAD, 16, panel + (w * 16) * S_PAD, S_PAD, 8, acc);
        store_c_bf16(attnout + cg, S_PAD, acc, 16);
        __syncthreads();
    }

    // ---- Phase 5: x = x_anc + attnout @ W_o ----
#pragma unroll
    for (int ch = 0; ch < 2; ++ch) {
        stageT(panel, Wo, DD, ch * 128);
        __syncthreads();
        v8f acc = {0.f, 0.f, 0.f, 0.f, 0.f, 0.f, 0.f, 0.f};
        acc = gemm16(attnout, S_PAD, 16, panel + (w * 16) * S_PAD, S_PAD, 8, acc);
        int n = lane & 15, half = lane >> 4;
        int cg = ch * 128 + w * 16 + n;
#pragma unroll
        for (int v = 0; v < 8; ++v) {
            int m = v + 8 * half;
            float r = xa[(size_t)(b0 + m) * DD + cg];
            xout[(size_t)(b0 + m) * DD + cg] = acc[v] + r;
        }
        __syncthreads();
    }
}

// =====================================================================================
// Kernel B: fused LN2 + FF1(+bias, exact GELU) + FF2(+bias) + residual
// =====================================================================================
__global__ void __launch_bounds__(256)
ffn_kernel(const float* __restrict__ xin, const float* __restrict__ g2, const float* __restrict__ b2,
           const float* __restrict__ W1, const float* __restrict__ bias1,
           const float* __restrict__ W2, const float* __restrict__ bias2,
           float* __restrict__ out) {
    extern __shared__ char smem[];
    float*    xres  = (float*)smem;                    // [16][S_PAD] residual
    uint16_t* hb    = (uint16_t*)(xres + 16 * S_PAD);  // [16][S_PAD] ln2(x)
    uint16_t* gb    = hb + 16 * S_PAD;                 // [16][GPAD]  gelu activations
    uint16_t* panel = gb + 16 * GPAD;                  // [128][S_PAD]

    int b0 = blockIdx.x * 16;
    int tid = threadIdx.x, lane = tid & 31, w = tid >> 5;

#if defined(HAVE_ASYNC_LDS)
    // Async DMA the residual tile (pure f32 copy, no conversion) global -> LDS,
    // then LayerNorm reads it back from LDS. Uses ASYNCcnt, not LOADcnt.
    {
        // 16 rows x 256 f32 = 1024 x b128 transfers; 256 threads -> 4 rounds
        for (int i = tid; i < 16 * (DD / 4); i += 256) {
            int r = i >> 6;          // /64 vec4s per row
            int c = (i & 63) * 4;    // float col
            const float* gsrc = xin + (size_t)(b0 + r) * DD + c;
            float* ldst = xres + r * S_PAD + c;
            __builtin_amdgcn_global_load_async_to_lds_b128(
                (as1_v4i_ptr)(void*)gsrc, (as3_v4i_ptr)(void*)ldst, 0, 0);
        }
        __builtin_amdgcn_s_wait_asynccnt(0);
        __syncthreads();
        int r = 2 * w + (lane >> 4);
        ln_row(xres + r * S_PAD, g2, b2, hb + r * S_PAD, nullptr);
    }
#else
    {
        int r = 2 * w + (lane >> 4);
        ln_row(xin + (size_t)(b0 + r) * DD, g2, b2, hb + r * S_PAD, xres + r * S_PAD);
    }
#endif
    __syncthreads();

    // FF1: [16,256] @ [256,1024], 8 chunks of 128 output cols, + bias + exact gelu
#pragma unroll
    for (int ch = 0; ch < 8; ++ch) {
        stageT(panel, W1, FFD, ch * 128);
        __syncthreads();
        int cg = ch * 128 + w * 16;
        v8f acc = {0.f, 0.f, 0.f, 0.f, 0.f, 0.f, 0.f, 0.f};
        acc = gemm16(hb, S_PAD, 16, panel + (w * 16) * S_PAD, S_PAD, 8, acc);
        int n = lane & 15, half = lane >> 4;
        float bs = bias1[cg + n];
#pragma unroll
        for (int v = 0; v < 8; ++v) {
            int m = v + 8 * half;
            float x = acc[v] + bs;
            float g = 0.5f * x * (1.0f + erff(x * 0.70710678118654752f));
            gb[m * GPAD + cg + n] = f2bf(g);
        }
        __syncthreads();
    }

    // FF2: [16,1024] @ [1024,256], 2 output chunks x 4 K-sections, + bias + residual
#pragma unroll
    for (int ch = 0; ch < 2; ++ch) {
        v8f acc = {0.f, 0.f, 0.f, 0.f, 0.f, 0.f, 0.f, 0.f};
#pragma unroll
        for (int kc = 0; kc < 4; ++kc) {
            stageT(panel, W2 + (size_t)(kc * 256) * DD, DD, ch * 128);
            __syncthreads();
            acc = gemm16(gb + kc * 256, GPAD, 16, panel + (w * 16) * S_PAD, S_PAD, 8, acc);
            __syncthreads();
        }
        int n = lane & 15, half = lane >> 4;
        int cg = ch * 128 + w * 16 + n;
        float bs = bias2[cg];
#pragma unroll
        for (int v = 0; v < 8; ++v) {
            int m = v + 8 * half;
            out[(size_t)(b0 + m) * DD + cg] = acc[v] + bs + xres[m * S_PAD + cg];
        }
    }
}

// =====================================================================================

extern "C" void kernel_launch(void* const* d_in, const int* in_sizes, int n_in,
                              void* d_out, int out_size, void* d_ws, size_t ws_size,
                              hipStream_t stream) {
    const float* xa    = (const float*)d_in[0];
    const float* xn    = (const float*)d_in[1];
    const float* Wq    = (const float*)d_in[2];
    const float* Wk    = (const float*)d_in[3];
    const float* Wv    = (const float*)d_in[4];
    const float* Wo    = (const float*)d_in[5];
    const float* g1    = (const float*)d_in[6];
    const float* b1    = (const float*)d_in[7];
    const float* g2    = (const float*)d_in[8];
    const float* b2    = (const float*)d_in[9];
    const float* W1    = (const float*)d_in[10];
    const float* bias1 = (const float*)d_in[11];
    const float* W2    = (const float*)d_in[12];
    const float* bias2 = (const float*)d_in[13];
    float* outp = (float*)d_out;
    float* xmid = (float*)d_ws;

    size_t smemA = (size_t)(16 * S_PAD) * 2      // lnx
                 + (size_t)(16 * S_PAD) * 2      // Qbf
                 + (size_t)(16 * 8 * S_PAD) * 2  // qc
                 + (size_t)(8 * 16 * 40) * 2     // attnb
                 + (size_t)(8 * 32 * S_PAD) * 2; // union (xn) >= panel
    size_t smemB = (size_t)(16 * S_PAD) * 4      // xres
                 + (size_t)(16 * S_PAD) * 2      // hb
                 + (size_t)(16 * GPAD) * 2       // gb
                 + (size_t)(128 * S_PAD) * 2;    // panel

    (void)hipFuncSetAttribute((const void*)attn_kernel,
                              hipFuncAttributeMaxDynamicSharedMemorySize, (int)smemA);
    (void)hipFuncSetAttribute((const void*)ffn_kernel,
                              hipFuncAttributeMaxDynamicSharedMemorySize, (int)smemB);

    attn_kernel<<<BB / 16, 256, smemA, stream>>>(xa, xn, Wq, Wk, Wv, Wo, g1, b1, xmid);
    ffn_kernel<<<BB / 16, 256, smemB, stream>>>(xmid, g2, b2, W1, bias1, W2, bias2, outp);
}
